// RevGNN_4071628996858
// MI455X (gfx1250) — matrix operand order repaired and checked by reading, not compiled, not run on hardware
//
#include <hip/hip_runtime.h>

// ---------------------------------------------------------------------------
// RevGNN forward for MI455X (gfx1250, wave32, WMMA).
//  * All dense GEMMs: v_wmma_f32_16x16x32_bf16, operands pre-packed to
//    fragment-major bf16 so the inner loop is pure b128 loads + wmma.
//  * Weight tiles staged into LDS via GLOBAL_LOAD_ASYNC_TO_LDS_B128
//    (CDNA5 async path, ASYNCcnt) instead of a VGPR round-trip.
//  * Edge aggregation: L2-resident coalesced float atomics + global prefetch.
// ---------------------------------------------------------------------------

#define NNODES  100000
#define INFEAT  100
#define HFEAT   128
#define OUTFEAT 47
#define NLAYERS 2
#define NGROUPS 2
#define DG      64
#define LN_EPS  1e-5f

typedef __attribute__((ext_vector_type(16))) __bf16 v16bf;
typedef __attribute__((ext_vector_type(8)))  float  v8f;

union FragLd { v16bf v; uint4 q[2]; };

__device__ __forceinline__ unsigned int pack2_bf16(float a, float b) {
    unsigned int ua = __float_as_uint(a);
    ua += 0x7FFFu + ((ua >> 16) & 1u);                    // RNE
    unsigned int ub = __float_as_uint(b);
    ub += 0x7FFFu + ((ub >> 16) & 1u);
    return (ua >> 16) | (ub & 0xFFFF0000u);
}

// ---------------------------------------------------------------------------
// Async global->LDS staging of a packed-B tile block (ASYNCcnt path).
// Each enabled lane copies 16B per iteration directly into LDS, no VGPR data.
// LDS byte address = low 32 bits of the generic shared pointer.
__device__ __forceinline__ void stage_b_async(unsigned int* ldsDst,
                                              const unsigned int* gSrc,
                                              int nWords) {
    for (int i = threadIdx.x * 4; i < nWords; i += blockDim.x * 4) {
        unsigned lds_addr = (unsigned)(unsigned long long)(const void*)(ldsDst + i);
        asm volatile("global_load_async_to_lds_b128 %0, %1, off"
                     :: "v"(lds_addr), "v"(gSrc + i)
                     : "memory");
    }
    asm volatile("s_wait_asynccnt 0" ::: "memory");
    __syncthreads();
}

// ---------------------------------------------------------------------------
// Pack a row-major f32 matrix A[rows x lda] into A-fragment-major bf16:
//   word index = ((tile*nSteps + step)*32 + lane)*8 + v
// ISA 7.12.2 A 16x32 layout: m = lane&15; v<4 -> K = 32*step + half*8 + 2v+{0,1};
//                            v>=4 -> K = 32*step + 16 + half*8 + 2(v-4)+{0,1}.
// Optional mean-scale: scale = 1/max(deg[row],1).  K clamped+zeroed past kmax.
__global__ void pack_a_kernel(const float* __restrict__ A, int lda, int kmax,
                              int nSteps, int nTiles,
                              const float* __restrict__ degv,
                              unsigned int* __restrict__ out) {
    int wave = (int)(((long long)blockIdx.x * blockDim.x + threadIdx.x) >> 5);
    if (wave >= nTiles * nSteps) return;
    int step = wave % nSteps;
    int lane = threadIdx.x & 31;
    int half = lane >> 4, m = lane & 15;
    int row  = (wave / nSteps) * 16 + m;
    float scale = degv ? (1.f / fmaxf(degv[row], 1.f)) : 1.f;
    const float* rp = A + (size_t)row * lda;
    unsigned int w[8];
#pragma unroll
    for (int v = 0; v < 8; ++v) {
        int kb = (v < 4) ? (step * 32 + half * 8 + 2 * v)
                         : (step * 32 + 16 + half * 8 + 2 * (v - 4));
        int k0 = kb, k1 = kb + 1;
        float f0 = rp[k0 < kmax ? k0 : kmax - 1];         // clamped load, no branch
        float f1 = rp[k1 < kmax ? k1 : kmax - 1];
        f0 = (k0 < kmax) ? f0 * scale : 0.f;              // v_cndmask
        f1 = (k1 < kmax) ? f1 * scale : 0.f;
        w[v] = pack2_bf16(f0, f1);
    }
    uint4* op = (uint4*)(out + ((size_t)wave * 32 + lane) * 8);
    op[0] = make_uint4(w[0], w[1], w[2], w[3]);
    op[1] = make_uint4(w[4], w[5], w[6], w[7]);
}

// Pack W[jmax x ldw] (row-major) as B = W^T fragments:
//   word index = ((colTile*nSteps + step)*32 + lane)*8 + v
// ISA B 32x16 layout: n = lane&15; K = 32*step + (v>>2)*16 + half*8 + 2(v&3)+{0,1}.
// Rows >= jmax and K >= kmax produce zeros (clamped addresses, value select).
__global__ void pack_b_kernel(const float* __restrict__ W, int ldw, int kmax,
                              int jmax, int nSteps, int colTiles,
                              unsigned int* __restrict__ out) {
    int wave = (int)(((long long)blockIdx.x * blockDim.x + threadIdx.x) >> 5);
    if (wave >= colTiles * nSteps) return;
    int step = wave % nSteps;
    int colT = wave / nSteps;
    int lane = threadIdx.x & 31;
    int half = lane >> 4, n = lane & 15;
    int j = colT * 16 + n;
    float rowScale = (j < jmax) ? 1.f : 0.f;
    const float* rp = W + (size_t)(j < jmax ? j : jmax - 1) * ldw;
    unsigned int w[8];
#pragma unroll
    for (int v = 0; v < 8; ++v) {
        int k0 = step * 32 + (v >> 2) * 16 + half * 8 + 2 * (v & 3);
        int k1 = k0 + 1;
        float f0 = rp[k0 < kmax ? k0 : kmax - 1];
        float f1 = rp[k1 < kmax ? k1 : kmax - 1];
        f0 = (k0 < kmax) ? f0 * rowScale : 0.f;
        f1 = (k1 < kmax) ? f1 * rowScale : 0.f;
        w[v] = pack2_bf16(f0, f1);
    }
    uint4* op = (uint4*)(out + ((size_t)wave * 32 + lane) * 8);
    op[0] = make_uint4(w[0], w[1], w[2], w[3]);
    op[1] = make_uint4(w[4], w[5], w[6], w[7]);
}

__device__ __forceinline__ v16bf load_frag(const unsigned int* pk,
                                           int tile, int nSteps, int step, int lane) {
    FragLd f;
    const uint4* p = (const uint4*)(pk + ((size_t)(tile * nSteps + step) * 32 + lane) * 8);
    f.q[0] = p[0];
    f.q[1] = p[1];
    return f.v;
}

// ---------------------------------------------------------------------------
__global__ void fill_kernel(float* __restrict__ p, float val, long long n) {
    long long i = (long long)blockIdx.x * blockDim.x + threadIdx.x;
    if (i < n) p[i] = val;
}

__global__ void degree_kernel(const int* __restrict__ dst, float* __restrict__ deg, long long E) {
    long long e = (long long)blockIdx.x * blockDim.x + threadIdx.x;
    if (e < E) atomicAdd(&deg[dst[e]], 1.0f);
}

// 16 lanes per edge, 4 channels each -> coalesced 64B atomic bursts into agg[dst].
// Prefetch the z-row of an edge ~128 ahead to hide gather latency.
__global__ void scatter_kernel(const int* __restrict__ src, const int* __restrict__ dst,
                               const float* __restrict__ z, float* __restrict__ agg,
                               long long E) {
    long long gid = (long long)blockIdx.x * blockDim.x + threadIdx.x;
    long long e   = gid >> 4;
    if (e >= E) return;
    int c0 = (int)(gid & 15) * 4;
    if ((gid & 15) == 0) {
        long long ef = e + 128;
        if (ef < E) __builtin_prefetch(z + (size_t)src[ef] * DG, 0, 1);
    }
    long long s = src[e], d = dst[e];
    const float4 v = *(const float4*)(z + s * DG + c0);
    float* ap = agg + d * DG + c0;
    atomicAdd(ap + 0, v.x);
    atomicAdd(ap + 1, v.y);
    atomicAdd(ap + 2, v.z);
    atomicAdd(ap + 3, v.w);
}

// One wave per node: LayerNorm over C channels (C in {64,128}) + ReLU.
__global__ void ln_relu_kernel(const float* __restrict__ in, int in_ld, int in_coff,
                               int C, const float* __restrict__ g, const float* __restrict__ b,
                               float* __restrict__ out, int out_ld, int N) {
    int node = (int)(((long long)blockIdx.x * blockDim.x + threadIdx.x) >> 5);
    int lane = threadIdx.x & 31;
    if (node >= N) return;
    const float* row = in + (size_t)node * in_ld + in_coff;
    float vals[4];
    int per = C >> 5;
    float s = 0.f;
#pragma unroll 4
    for (int i = 0; i < per; ++i) { vals[i] = row[lane + 32 * i]; s += vals[i]; }
#pragma unroll
    for (int off = 16; off; off >>= 1) s += __shfl_xor(s, off, 32);
    float mu = s / (float)C;
    float vs = 0.f;
#pragma unroll 4
    for (int i = 0; i < per; ++i) { float d = vals[i] - mu; vs += d * d; }
#pragma unroll
    for (int off = 16; off; off >>= 1) vs += __shfl_xor(vs, off, 32);
    float inv = rsqrtf(vs / (float)C + LN_EPS);
    float* orow = out + (size_t)node * out_ld;
#pragma unroll 4
    for (int i = 0; i < per; ++i) {
        int c = lane + 32 * i;
        float y = (vals[i] - mu) * inv * g[c] + b[c];
        orow[c] = fmaxf(y, 0.f);
    }
}

// ---------------------------------------------------------------------------
// out[row, col] = A @ W^T + bias  (packed bf16 operands; B async-staged in LDS).
// STEPS = K/32.  Column guard handles OUTFEAT=47.
template <int STEPS>
__global__ void gemm_packed_kernel(const unsigned int* __restrict__ Apk,
                                   const unsigned int* __restrict__ Bpk,
                                   const float* __restrict__ bias,
                                   float* __restrict__ out, int out_ld,
                                   int colTiles, int ncols) {
    extern __shared__ __align__(16) unsigned int ldsB[];
    stage_b_async(ldsB, Bpk, colTiles * STEPS * 32 * 8);

    int wave = (int)(((long long)blockIdx.x * blockDim.x + threadIdx.x) >> 5);
    int rowT = wave / colTiles, colT = wave % colTiles;
    if (rowT >= NNODES / 16) return;
    int lane = threadIdx.x & 31, half = lane >> 4, n = lane & 15;

    v8f acc = {};
#pragma unroll
    for (int s = 0; s < STEPS; ++s) {
        v16bf a = load_frag(Apk, rowT, STEPS, s, lane);
        v16bf b = load_frag(ldsB, colT, STEPS, s, lane);
        acc = __builtin_amdgcn_wmma_f32_16x16x32_bf16(false, a, false, b,
                                                      (short)0, acc, false, false);
    }
    int col = colT * 16 + n;
    if (col >= ncols) return;
    float bv = bias[col];
    int row0 = rowT * 16;
#pragma unroll
    for (int v = 0; v < 8; ++v) {
        int m = v + half * 8;
        out[(size_t)(row0 + m) * out_ld + col] = acc[v] + bv;
    }
}

// h[:, gofs + col] += mean @ Wl^T + bl + z @ Wr^T  (mean pre-scaled in aggPk).
// LDS holds WlPk then WrPk back to back.
__global__ void conv_gemm_kernel(const unsigned int* __restrict__ aggPk,
                                 const unsigned int* __restrict__ zPk,
                                 const unsigned int* __restrict__ WPk,  // Wl | Wr
                                 const float* __restrict__ bl,
                                 float* __restrict__ h, int gofs) {
    const int STEPS = DG / 32;              // 2
    const int colTiles = DG / 16;           // 4
    const int nwHalf = colTiles * STEPS * 32 * 8;
    extern __shared__ __align__(16) unsigned int ldsB[];
    stage_b_async(ldsB, WPk, 2 * nwHalf);

    int wave = (int)(((long long)blockIdx.x * blockDim.x + threadIdx.x) >> 5);
    int rowT = wave / colTiles, colT = wave % colTiles;
    if (rowT >= NNODES / 16) return;
    int lane = threadIdx.x & 31, half = lane >> 4, n = lane & 15;

    v8f acc = {};
#pragma unroll
    for (int s = 0; s < STEPS; ++s) {
        v16bf am  = load_frag(aggPk, rowT, STEPS, s, lane);
        v16bf bwl = load_frag(ldsB, colT, STEPS, s, lane);
        acc = __builtin_amdgcn_wmma_f32_16x16x32_bf16(false, am, false, bwl,
                                                      (short)0, acc, false, false);
        v16bf az  = load_frag(zPk, rowT, STEPS, s, lane);
        v16bf bwr = load_frag(ldsB + nwHalf, colT, STEPS, s, lane);
        acc = __builtin_amdgcn_wmma_f32_16x16x32_bf16(false, az, false, bwr,
                                                      (short)0, acc, false, false);
    }
    float bv = bl[colT * 16 + n];
    int row0 = rowT * 16;
#pragma unroll
    for (int v = 0; v < 8; ++v) {
        int m = v + half * 8;
        float* p = h + (size_t)(row0 + m) * HFEAT + gofs + colT * 16 + n;
        *p += acc[v] + bv;
    }
}

// ---------------------------------------------------------------------------
extern "C" void kernel_launch(void* const* d_in, const int* in_sizes, int n_in,
                              void* d_out, int out_size, void* d_ws, size_t ws_size,
                              hipStream_t stream) {
    const float* x      = (const float*)d_in[0];
    const int*   ei     = (const int*)d_in[1];
    const float* lin1_w = (const float*)d_in[2];
    const float* lin1_b = (const float*)d_in[3];
    const float* lin2_w = (const float*)d_in[4];
    const float* lin2_b = (const float*)d_in[5];
    const float* norm_g = (const float*)d_in[6];
    const float* norm_b = (const float*)d_in[7];
    const float* ln_g   = (const float*)d_in[8];
    const float* ln_b   = (const float*)d_in[9];
    const float* Wl     = (const float*)d_in[10];
    const float* bl     = (const float*)d_in[11];
    const float* Wr     = (const float*)d_in[12];
    float*       out    = (float*)d_out;

    const long long E = (long long)in_sizes[1] / 2;
    const int* srcI = ei;
    const int* dstI = ei + E;

    // ---- workspace layout -------------------------------------------------
    float* h    = (float*)d_ws;                            // N*128 f32
    float* z    = h   + (size_t)NNODES * HFEAT;            // N*64  f32
    float* agg  = z   + (size_t)NNODES * DG;               // N*64  f32
    float* degv = agg + (size_t)NNODES * DG;               // N     f32
    float* tbuf = z;                                       // N*128 f32 (reuses z+agg)
    unsigned int* aPkBig = (unsigned int*)(degv + NNODES); // N*128 bf16 = N*64 u32
    unsigned int* zPk    = aPkBig + (size_t)NNODES * 64;   // N*64 bf16 = N*32 u32
    unsigned int* aggPk  = zPk    + (size_t)NNODES * 32;   // N*32 u32
    unsigned int* wPk1   = aggPk  + (size_t)NNODES * 32;   // 8*4*32*8   = 8192 u32
    unsigned int* wPk2   = wPk1 + 8 * 4 * 32 * 8;          // 3*4*32*8   = 3072 u32
    unsigned int* wPkCv  = wPk2 + 3 * 4 * 32 * 8;          // 4*(2*2048) = 16384 u32
    const int convWords  = (DG / 16) * (DG / 32) * 32 * 8; // 2048 u32 per 64x64 W

    const int TB = 256;
    const int nodeTiles = NNODES / 16;
    auto wgrid = [&](long long waves) { return (int)((waves * 32 + TB - 1) / TB); };

    // ---- degrees (dst shared by all convs) --------------------------------
    fill_kernel<<<(NNODES + TB - 1) / TB, TB, 0, stream>>>(degv, 0.f, NNODES);
    degree_kernel<<<(int)((E + TB - 1) / TB), TB, 0, stream>>>(dstI, degv, E);

    // ---- pack all weights (tiny) ------------------------------------------
    pack_b_kernel<<<wgrid(8 * 4), TB, 0, stream>>>(lin1_w, INFEAT, INFEAT, HFEAT, 4, 8, wPk1);
    pack_b_kernel<<<wgrid(3 * 4), TB, 0, stream>>>(lin2_w, HFEAT, HFEAT, OUTFEAT, 4, 3, wPk2);
    for (int i = 0; i < NLAYERS * NGROUPS; ++i) {
        pack_b_kernel<<<wgrid(4 * 2), TB, 0, stream>>>(
            Wl + (size_t)i * DG * DG, DG, DG, DG, 2, 4, wPkCv + (size_t)(2 * i) * convWords);
        pack_b_kernel<<<wgrid(4 * 2), TB, 0, stream>>>(
            Wr + (size_t)i * DG * DG, DG, DG, DG, 2, 4, wPkCv + (size_t)(2 * i + 1) * convWords);
    }

    // ---- h = x @ lin1_w^T + b ---------------------------------------------
    pack_a_kernel<<<wgrid((long long)nodeTiles * 4), TB, 0, stream>>>(
        x, INFEAT, INFEAT, 4, nodeTiles, nullptr, aPkBig);
    gemm_packed_kernel<4><<<wgrid((long long)nodeTiles * 8), TB, 8 * 4 * 1024, stream>>>(
        aPkBig, wPk1, lin1_b, h, HFEAT, 8, HFEAT);

    // ---- reversible layers -------------------------------------------------
    for (int l = 0; l < NLAYERS; ++l) {
        for (int g = 0; g < NGROUPS; ++g) {
            int in_off = (g == 0) ? DG : 0;
            size_t pi = (size_t)l * NGROUPS + g;
            ln_relu_kernel<<<wgrid(NNODES), TB, 0, stream>>>(
                h, HFEAT, in_off, DG, ln_g + pi * DG, ln_b + pi * DG, z, DG, NNODES);
            fill_kernel<<<(int)(((long long)NNODES * DG + TB - 1) / TB), TB, 0, stream>>>(
                agg, 0.f, (long long)NNODES * DG);
            scatter_kernel<<<(int)((E * 16 + TB - 1) / TB), TB, 0, stream>>>(srcI, dstI, z, agg, E);
            // pack z and mean(agg) (mean fused via degv scale)
            pack_a_kernel<<<wgrid((long long)nodeTiles * 2), TB, 0, stream>>>(
                z, DG, DG, 2, nodeTiles, nullptr, zPk);
            pack_a_kernel<<<wgrid((long long)nodeTiles * 2), TB, 0, stream>>>(
                agg, DG, DG, 2, nodeTiles, degv, aggPk);
            conv_gemm_kernel<<<wgrid((long long)nodeTiles * 4), TB, 2 * convWords * 4, stream>>>(
                aggPk, zPk, wPkCv + 2 * pi * convWords, bl + pi * DG, h, g * DG);
        }
    }

    // ---- final LN + lin2 ---------------------------------------------------
    ln_relu_kernel<<<wgrid(NNODES), TB, 0, stream>>>(
        h, HFEAT, 0, HFEAT, norm_g, norm_b, tbuf, HFEAT, NNODES);
    pack_a_kernel<<<wgrid((long long)nodeTiles * 4), TB, 0, stream>>>(
        tbuf, HFEAT, HFEAT, 4, nodeTiles, nullptr, aPkBig);
    gemm_packed_kernel<4><<<wgrid((long long)nodeTiles * 3), TB, 3 * 4 * 1024, stream>>>(
        aPkBig, wPk2, lin2_b, out, OUTFEAT, 3, OUTFEAT);
}
